// TemporalAttention_51273319580274
// MI455X (gfx1250) — matrix operand order repaired and checked by reading, not compiled, not run on hardware
//
#include <hip/hip_runtime.h>

// Problem constants (from reference)
#define BB   4
#define TT   256
#define RR   100
#define DD   128
#define NSEQ (BB * RR)   // 400 sequences

typedef __bf16 bf16;
typedef __attribute__((ext_vector_type(16))) __bf16 v16bf;
typedef __attribute__((ext_vector_type(8)))  __bf16 v8bf;
typedef __attribute__((ext_vector_type(4)))  __bf16 v4bf;
typedef __attribute__((ext_vector_type(8)))  float  v8f;

union V16U { v16bf v; v8bf h[2]; };

// ---- WMMA operand loaders (CDNA5 ISA 7.12.2 layouts, wave32) ----
// A (16x32 bf16): lane<16 row=lane%16 holds K[0..7],K[16..23]; lane>=16 holds K[8..15],K[24..31]
__device__ __forceinline__ v16bf ldsA(const bf16* base, int ld, int rowBase, int kBase, int lane) {
  int r  = rowBase + (lane & 15);
  int ko = (lane & 16) ? 8 : 0;
  const bf16* p = base + r * ld + kBase + ko;
  V16U u;
  u.h[0] = *(const v8bf*)(p);
  u.h[1] = *(const v8bf*)(p + 16);
  return u.v;
}
// B (32x16 bf16) from B^T stored row-major [N][K]: lane<16 col=lane holds K[0..7],K[16..23]; lane>=16 K[8..15],K[24..31]
__device__ __forceinline__ v16bf ldsB(const bf16* bt, int ld, int colBase, int kBase, int lane) {
  int c  = colBase + (lane & 15);
  int ko = (lane & 16) ? 8 : 0;
  const bf16* p = bt + c * ld + kBase + ko;
  V16U u;
  u.h[0] = *(const v8bf*)(p);
  u.h[1] = *(const v8bf*)(p + 16);
  return u.v;
}

__device__ __forceinline__ v8f wmma_bf16(v16bf a, v16bf b, v8f c) {
  // D = A*B + C, f32 accumulate. 8-arg form: (neg_a, A, neg_b, B, c_mod, C, reuse_a, reuse_b)
  return __builtin_amdgcn_wmma_f32_16x16x32_bf16(false, a, false, b, (short)0, c, false, false);
}

// LDS layout (bytes):
//   [0      , 65536 ) Xl  bf16[256][128]      -> later P scratch (8KB per wave)
//   [65536  , 98304 ) WqT bf16[128][128]
//   [98304  ,131072 ) WkT bf16[128][128]      -> later WpT
//   [131072 ,163840 ) WvT bf16[128][128]      -> later Ql bf16[128][128]
//   [163840 ,229376 ) Kl  bf16[256][128]
//   [229376 ,294912 ) Vtl bf16[128][256]      (V transposed)
//   [294912 ,...    ) tables: bq|bk|bv|bp|g|b (6*128 f32), rel (512 f32), mask (256 i32)
#define SMEM_BYTES (294912 + 768*4 + 512*4 + 256*4)

__global__ void __launch_bounds__(256)
temporal_attention_wmma_kernel(
    const float* __restrict__ x,   const int*   __restrict__ mask,
    const float* __restrict__ Wq,  const float* __restrict__ bq,
    const float* __restrict__ Wk,  const float* __restrict__ bk,
    const float* __restrict__ Wv,  const float* __restrict__ bv,
    const float* __restrict__ Wp,  const float* __restrict__ bp,
    const float* __restrict__ lng, const float* __restrict__ lnb,
    const float* __restrict__ rel, float* __restrict__ out)
{
  extern __shared__ __align__(16) char smem[];
  bf16*  Xl   = (bf16*)(smem);
  bf16*  WqT  = (bf16*)(smem + 65536);
  bf16*  WkT  = (bf16*)(smem + 98304);
  bf16*  WvT  = (bf16*)(smem + 131072);
  bf16*  Kl   = (bf16*)(smem + 163840);
  bf16*  Vtl  = (bf16*)(smem + 229376);
  float* tb   = (float*)(smem + 294912);   // bq@0 bk@128 bv@256 bp@384 g@512 b@640
  float* relS = tb + 768;                  // 511 used
  int*   mk   = (int*)(relS + 512);        // 256

  const int tid  = threadIdx.x;
  const int lane = tid & 31;
  const int wv   = tid >> 5;               // 8 waves
  const int n    = blockIdx.x >> 1;        // sequence
  const int qblk = blockIdx.x & 1;         // which 128-row query half
  const float* __restrict__ xs = x + (size_t)n * TT * DD;

  const int lr0  = (lane & 16) ? 8 : 0;    // C-layout row offset for this lane half
  const int qrow = qblk * 128 + wv * 16;   // this wave's global query row base

  // warm the residual rows in L2 (global_prefetch_b8)
  __builtin_prefetch(xs + (size_t)qrow * DD, 0, 1);

  // ---- Stage x -> bf16 LDS (coalesced float4) ----
  for (int it = 0; it < 32; ++it) {
    int idx = tid + it * 256;                       // float4 index, 8192 total
    float4 vx = ((const float4*)xs)[idx];
    v4bf o; o[0] = (bf16)vx.x; o[1] = (bf16)vx.y; o[2] = (bf16)vx.z; o[3] = (bf16)vx.w;
    *((v4bf*)Xl + idx) = o;
  }
  // ---- Stage W^T (bf16, transposed so B-columns are contiguous) ----
  for (int it = 0; it < 64; ++it) {
    int idx = tid + it * 256; int k = idx >> 7, cc = idx & 127;
    WqT[cc * DD + k] = (bf16)Wq[idx];
    WkT[cc * DD + k] = (bf16)Wk[idx];
    WvT[cc * DD + k] = (bf16)Wv[idx];
  }
  if (tid < 128) {
    tb[tid]       = bq[tid];  tb[128 + tid] = bk[tid];  tb[256 + tid] = bv[tid];
    tb[384 + tid] = bp[tid];  tb[512 + tid] = lng[tid]; tb[640 + tid] = lnb[tid];
  }
  for (int i = tid; i < 511; i += 256) relS[i] = rel[i];
  if (tid < 256) mk[tid] = mask[(n / RR) * TT + tid];
  __syncthreads();

  // ---- K = X @ Wk + bk  (16 M-tiles over 8 waves) ----
  for (int m = wv; m < 16; m += 8) {
    int rb = m * 16;
    v16bf a[4];
    #pragma unroll
    for (int kk = 0; kk < 4; ++kk) a[kk] = ldsA(Xl, DD, rb, kk * 32, lane);
    #pragma unroll
    for (int nt = 0; nt < 8; ++nt) {
      v8f c = {};
      #pragma unroll
      for (int kk = 0; kk < 4; ++kk) c = wmma_bf16(a[kk], ldsB(WkT, DD, nt * 16, kk * 32, lane), c);
      int col = nt * 16 + (lane & 15);
      float bias = tb[128 + col];
      #pragma unroll
      for (int i = 0; i < 8; ++i) Kl[(rb + lr0 + i) * DD + col] = (bf16)(c[i] + bias);
    }
  }
  // ---- V = X @ Wv + bv, stored transposed Vtl[d][s] ----
  for (int m = wv; m < 16; m += 8) {
    int rb = m * 16;
    v16bf a[4];
    #pragma unroll
    for (int kk = 0; kk < 4; ++kk) a[kk] = ldsA(Xl, DD, rb, kk * 32, lane);
    #pragma unroll
    for (int nt = 0; nt < 8; ++nt) {
      v8f c = {};
      #pragma unroll
      for (int kk = 0; kk < 4; ++kk) c = wmma_bf16(a[kk], ldsB(WvT, DD, nt * 16, kk * 32, lane), c);
      int col = nt * 16 + (lane & 15);
      float bias = tb[256 + col];
      #pragma unroll
      for (int i = 0; i < 8; ++i) Vtl[col * TT + (rb + lr0 + i)] = (bf16)(c[i] + bias);
    }
  }
  __syncthreads();   // WkT/WvT dead for all waves after this

  // ---- Q = X @ Wq + bq for this wave's 16 rows -> Ql (reuses WvT region) ----
  bf16* Ql = WvT;
  {
    v16bf a[4];
    #pragma unroll
    for (int kk = 0; kk < 4; ++kk) a[kk] = ldsA(Xl, DD, qrow, kk * 32, lane);
    #pragma unroll
    for (int nt = 0; nt < 8; ++nt) {
      v8f c = {};
      #pragma unroll
      for (int kk = 0; kk < 4; ++kk) c = wmma_bf16(a[kk], ldsB(WqT, DD, nt * 16, kk * 32, lane), c);
      int col = nt * 16 + (lane & 15);
      float bias = tb[col];
      #pragma unroll
      for (int i = 0; i < 8; ++i) Ql[(wv * 16 + lr0 + i) * DD + col] = (bf16)(c[i] + bias);
    }
  }
  // ---- Stage Wp^T into the dead WkT region ----
  bf16* WpT = WkT;
  for (int it = 0; it < 64; ++it) {
    int idx = tid + it * 256; int k = idx >> 7, cc = idx & 127;
    WpT[cc * DD + k] = (bf16)Wp[idx];
  }
  __syncthreads();   // Xl dead for all waves; becomes per-wave P scratch

  bf16* Pw = Xl + wv * (16 * TT);          // this wave's 8KB scratch

  // ---- S = Q K^T * scale + rel + masks (16 col-tiles, kept in registers) ----
  const float scale = 0.08838834764831845f;  // 1/sqrt(128)
  float s[16][8];
  {
    v16bf a[4];
    #pragma unroll
    for (int kk = 0; kk < 4; ++kk) a[kk] = ldsA(Ql, DD, wv * 16, kk * 32, lane);
    #pragma unroll
    for (int ct = 0; ct < 16; ++ct) {
      v8f c = {};
      #pragma unroll
      for (int kk = 0; kk < 4; ++kk) c = wmma_bf16(a[kk], ldsB(Kl, DD, ct * 16, kk * 32, lane), c);
      int col = ct * 16 + (lane & 15);
      int kmask = mk[col];
      #pragma unroll
      for (int i = 0; i < 8; ++i) {
        int rg  = qrow + lr0 + i;
        int rid = rg - col + 255;  rid = rid < 0 ? 0 : (rid > 510 ? 510 : rid);
        float v = c[i] * scale + relS[rid];
        if (kmask == 0) v = -1e9f;
        if (mk[rg] == 0) v = -1e9f;
        if (col > rg)   v = -1e9f;
        s[ct][i] = v;
      }
    }
  }
  // ---- softmax over 256 cols (16 lanes x 16 tiles); width-16 butterflies ----
  float mx[8], sm[8];
  #pragma unroll
  for (int i = 0; i < 8; ++i) {
    float m = -3e38f;
    #pragma unroll
    for (int ct = 0; ct < 16; ++ct) m = fmaxf(m, s[ct][i]);
    mx[i] = m;
  }
  #pragma unroll
  for (int off = 1; off < 16; off <<= 1)
    #pragma unroll
    for (int i = 0; i < 8; ++i) mx[i] = fmaxf(mx[i], __shfl_xor(mx[i], off, 16));
  #pragma unroll
  for (int i = 0; i < 8; ++i) sm[i] = 0.f;
  #pragma unroll
  for (int ct = 0; ct < 16; ++ct)
    #pragma unroll
    for (int i = 0; i < 8; ++i) { float e = __expf(s[ct][i] - mx[i]); s[ct][i] = e; sm[i] += e; }
  #pragma unroll
  for (int off = 1; off < 16; off <<= 1)
    #pragma unroll
    for (int i = 0; i < 8; ++i) sm[i] += __shfl_xor(sm[i], off, 16);
  #pragma unroll
  for (int i = 0; i < 8; ++i) sm[i] = 1.0f / sm[i];
  // P (bf16) -> per-wave LDS scratch, row-major [16][256]
  #pragma unroll
  for (int ct = 0; ct < 16; ++ct) {
    int col = ct * 16 + (lane & 15);
    #pragma unroll
    for (int i = 0; i < 8; ++i) Pw[(lr0 + i) * TT + col] = (bf16)(s[ct][i] * sm[i]);
  }

  // ---- ctx = P @ V  (K = 256, B from Vtl[d][s]) ----
  v8f ctx[8];
  {
    v16bf a[8];
    #pragma unroll
    for (int kk = 0; kk < 8; ++kk) a[kk] = ldsA(Pw, TT, 0, kk * 32, lane);
    #pragma unroll
    for (int nt = 0; nt < 8; ++nt) {
      v8f c = {};
      #pragma unroll
      for (int kk = 0; kk < 8; ++kk) c = wmma_bf16(a[kk], ldsB(Vtl, TT, nt * 16, kk * 32, lane), c);
      ctx[nt] = c;
    }
  }
  // ctx -> bf16 A-layout via own scratch [16][128]
  #pragma unroll
  for (int nt = 0; nt < 8; ++nt) {
    int col = nt * 16 + (lane & 15);
    #pragma unroll
    for (int i = 0; i < 8; ++i) Pw[(lr0 + i) * DD + col] = (bf16)ctx[nt][i];
  }

  // ---- out = ctx @ Wp + bp + x (residual) ----
  float y[8][8];
  {
    v16bf a[4];
    #pragma unroll
    for (int kk = 0; kk < 4; ++kk) a[kk] = ldsA(Pw, DD, 0, kk * 32, lane);
    #pragma unroll
    for (int nt = 0; nt < 8; ++nt) {
      v8f c = {};
      #pragma unroll
      for (int kk = 0; kk < 4; ++kk) c = wmma_bf16(a[kk], ldsB(WpT, DD, nt * 16, kk * 32, lane), c);
      int col = nt * 16 + (lane & 15);
      float bias = tb[384 + col];
      #pragma unroll
      for (int i = 0; i < 8; ++i) {
        int rg = qrow + lr0 + i;
        y[nt][i] = c[i] + bias + xs[(size_t)rg * DD + col];
      }
    }
  }
  // ---- LayerNorm over D = 16 lanes x 8 tiles ----
  float su[8], sq[8];
  #pragma unroll
  for (int i = 0; i < 8; ++i) {
    float a = 0.f, b2 = 0.f;
    #pragma unroll
    for (int nt = 0; nt < 8; ++nt) { float v = y[nt][i]; a += v; b2 += v * v; }
    su[i] = a; sq[i] = b2;
  }
  #pragma unroll
  for (int off = 1; off < 16; off <<= 1)
    #pragma unroll
    for (int i = 0; i < 8; ++i) { su[i] += __shfl_xor(su[i], off, 16); sq[i] += __shfl_xor(sq[i], off, 16); }
  #pragma unroll
  for (int i = 0; i < 8; ++i) {
    float mu   = su[i] * (1.0f / 128.0f);
    float var  = sq[i] * (1.0f / 128.0f) - mu * mu;
    su[i] = mu;
    sq[i] = rsqrtf(var + 1e-5f);
  }
  // ---- store: out[b][t][r][d], n = b*R + r ----
  const int bI = n / RR, rI = n % RR;
  #pragma unroll
  for (int nt = 0; nt < 8; ++nt) {
    int col = nt * 16 + (lane & 15);
    float g = tb[512 + col], bb = tb[640 + col];
    #pragma unroll
    for (int i = 0; i < 8; ++i) {
      int rg = qrow + lr0 + i;
      size_t o = (((size_t)bI * TT + rg) * RR + rI) * (size_t)DD + col;
      out[o] = (y[nt][i] - su[i]) * sq[i] * g + bb;
    }
  }
}

extern "C" void kernel_launch(void* const* d_in, const int* in_sizes, int n_in,
                              void* d_out, int out_size, void* d_ws, size_t ws_size,
                              hipStream_t stream) {
  (void)in_sizes; (void)n_in; (void)out_size; (void)d_ws; (void)ws_size;
  const float* x    = (const float*)d_in[0];
  const int*   mask = (const int*)  d_in[1];
  const float* Wq   = (const float*)d_in[2];
  const float* bq   = (const float*)d_in[3];
  const float* Wk   = (const float*)d_in[4];
  const float* bk   = (const float*)d_in[5];
  const float* Wv   = (const float*)d_in[6];
  const float* bv   = (const float*)d_in[7];
  const float* Wp   = (const float*)d_in[8];
  const float* bp   = (const float*)d_in[9];
  const float* lng  = (const float*)d_in[10];
  const float* lnb  = (const float*)d_in[11];
  const float* rel  = (const float*)d_in[12];
  float* out = (float*)d_out;

  dim3 grid(NSEQ * 2), block(256);
  temporal_attention_wmma_kernel<<<grid, block, SMEM_BYTES, stream>>>(
      x, mask, Wq, bq, Wk, bk, Wv, bv, Wp, bp, lng, lnb, rel, out);
}
